// BayesianDropoutLSTM_42468636623062
// MI455X (gfx1250) — matrix operand order repaired and checked by reading, not compiled, not run on hardware
//
#include <hip/hip_runtime.h>
#include <cstdint>

// ---------------------------------------------------------------------------
// Types for CDNA5 WMMA (wave32): v_wmma_f32_16x16x32_bf16
// ---------------------------------------------------------------------------
typedef __attribute__((ext_vector_type(16))) __bf16 v16bf;
typedef __attribute__((ext_vector_type(8)))  float  v8f;

#define NWG_LSTM 16
#define B_  64
#define S_  512
#define E_  256
#define H_  512
#define G_  2048   // 4*H
#define T_  48

// ---------------------------------------------------------------------------
// bf16 helpers (bit-level, RNE)
// ---------------------------------------------------------------------------
__device__ __forceinline__ unsigned short f2bf(float f) {
  unsigned u = __float_as_uint(f);
  unsigned r = (u + 0x7FFFu + ((u >> 16) & 1u)) >> 16;
  return (unsigned short)r;
}
__device__ __forceinline__ float bf2f(unsigned short h) {
  return __uint_as_float(((unsigned)h) << 16);
}
__device__ __forceinline__ float sigf(float x) { return 1.0f / (1.0f + expf(-x)); }

// ---------------------------------------------------------------------------
// Fragment assembly from two 16-byte chunks (addresses precomputed by caller;
// K-loop steps are compile-time byte offsets -> global_load_b128 w/ IOFFSET).
//
// A fragment (16x32 bf16, MxK) per-lane base for row-major [M][lda]:
//   base = A + (m0 + (lane&15))*lda + (lane>>4)*8
//   chunk0 @ base + kt*32, chunk1 @ base + kt*32 + 16   (elements)
// B fragment (32x16 bf16) from packed tiles [ntile][ktile][lane][16]:
//   base = Bp + (ntile*KT*32 + lane)*16
//   chunk0 @ base + kt*512, chunk1 @ base + kt*512 + 8  (elements)
// ---------------------------------------------------------------------------
__device__ __forceinline__ v16bf frag2(const unsigned short* __restrict__ p,
                                       unsigned off0, unsigned off1) {
  union { v16bf v; uint4 u[2]; } t;
  t.u[0] = *(const uint4*)(p + off0);
  t.u[1] = *(const uint4*)(p + off1);
  return t.v;
}

__device__ __forceinline__ v8f wmma_bf16(v16bf a, v16bf b, v8f c) {
  return __builtin_amdgcn_wmma_f32_16x16x32_bf16(false, a, false, b, (short)0, c,
                                                 false, false);
}

__device__ __forceinline__ const unsigned short* a_base(
    const unsigned short* __restrict__ A, unsigned lda, unsigned m0, unsigned lane) {
  return A + (size_t)(m0 + (lane & 15)) * lda + ((lane >> 4) << 3);
}
__device__ __forceinline__ const unsigned short* b_base(
    const unsigned short* __restrict__ Bp, unsigned KT, unsigned ntile, unsigned lane) {
  return Bp + ((((size_t)ntile * KT) << 5) + lane) * 16u;
}

// ---------------------------------------------------------------------------
// Kernel: pack f32 weight W[N][K] (row-major) into WMMA B-fragment tile order.
// out[((ntile*KT + ktile)*32 + lane)*16 + e] = bf16(W[n][k])
//   n = ntile*16 + (lane&15);  k = ktile*32 + (lane>>4)*8 + (e&7) + (e>>3)*16
// ---------------------------------------------------------------------------
__global__ void pack_b_kernel(const float* __restrict__ W,
                              unsigned short* __restrict__ out,
                              unsigned N, unsigned K) {
  unsigned idx = blockIdx.x * blockDim.x + threadIdx.x;
  if (idx >= N * K) return;
  unsigned e     = idx & 15u;
  unsigned lane  = (idx >> 4) & 31u;
  unsigned tp    = idx >> 9;            // ntile*KT + ktile
  unsigned KT    = K >> 5;
  unsigned ntile = tp / KT;
  unsigned ktile = tp - ntile * KT;
  unsigned n = ntile * 16u + (lane & 15u);
  unsigned k = ktile * 32u + ((lane >> 4) << 3) + (e & 7u) + ((e >> 3) << 4);
  out[idx] = f2bf(W[(size_t)n * K + k]);
}

// ---------------------------------------------------------------------------
// Kernel: embedding gather, fp32 -> bf16, row-major [B*S][E]
// ---------------------------------------------------------------------------
__global__ void embed_kernel(const int* __restrict__ x,
                             const float* __restrict__ emb,
                             unsigned short* __restrict__ xe) {
  unsigned idx = blockIdx.x * blockDim.x + threadIdx.x;   // over 32768*256
  unsigned m = idx >> 8;
  unsigned e = idx & 255u;
  xe[idx] = f2bf(emb[(size_t)(unsigned)x[m] * E_ + e]);
}

// ---------------------------------------------------------------------------
// Kernel: px = xe @ W_ih^T   (M=32768, N=2048, K=256), bf16 in/out, f32 acc
// WG: 8 waves, each wave = 16 rows x 64 cols (4 accumulators). Grid (256,32).
// Software-pipelined by one K-tile.
// ---------------------------------------------------------------------------
__global__ __launch_bounds__(256) void gemm_px_kernel(
    const unsigned short* __restrict__ xe,
    const unsigned short* __restrict__ WihP,
    unsigned short* __restrict__ px) {
  const unsigned wv   = threadIdx.x >> 5;
  const unsigned lane = threadIdx.x & 31u;
  const unsigned m0   = blockIdx.x * 128u + wv * 16u;
  const unsigned n0   = blockIdx.y * 64u;

  const unsigned short* aB = a_base(xe, E_, m0, lane);
  const unsigned short* bB[4];
#pragma unroll
  for (int j = 0; j < 4; ++j) bB[j] = b_base(WihP, 8, (n0 >> 4) + j, lane);

  v16bf a0 = frag2(aB, 0u, 16u);
  v16bf b0[4];
#pragma unroll
  for (int j = 0; j < 4; ++j) b0[j] = frag2(bB[j], 0u, 8u);

  v8f acc[4] = {};
#pragma unroll
  for (unsigned kt = 0; kt < 8; ++kt) {           // K = 256 = 8 * 32
    v16bf a1 = a0;
    v16bf b1[4];
#pragma unroll
    for (int j = 0; j < 4; ++j) b1[j] = b0[j];
    if (kt < 7) {                                  // prefetch next tile's frags
      a1 = frag2(aB, (kt + 1) * 32u, (kt + 1) * 32u + 16u);
#pragma unroll
      for (int j = 0; j < 4; ++j) b1[j] = frag2(bB[j], (kt + 1) * 512u, (kt + 1) * 512u + 8u);
    }
#pragma unroll
    for (int j = 0; j < 4; ++j) acc[j] = wmma_bf16(a0, b0[j], acc[j]);
    a0 = a1;
#pragma unroll
    for (int j = 0; j < 4; ++j) b0[j] = b1[j];
  }
  const unsigned half = lane >> 4, n15 = lane & 15u;
#pragma unroll
  for (int j = 0; j < 4; ++j)
#pragma unroll
    for (int r = 0; r < 8; ++r)
      px[(size_t)(m0 + half * 8u + r) * G_ + n0 + j * 16u + n15] = f2bf(acc[j][r]);
}

// ---------------------------------------------------------------------------
// Device-wide barrier for the persistent LSTM kernel (16 WGs).
// ---------------------------------------------------------------------------
__device__ __forceinline__ void grid_barrier(unsigned* cnt, unsigned& expected) {
  __threadfence();
  __syncthreads();
  if (threadIdx.x == 0) {
    atomicAdd(cnt, 1u);
    while (__hip_atomic_load(cnt, __ATOMIC_ACQUIRE, __HIP_MEMORY_SCOPE_AGENT) < expected)
      __builtin_amdgcn_s_sleep(2);
  }
  __syncthreads();
  expected += NWG_LSTM;
}

// ---------------------------------------------------------------------------
// Kernel: persistent LSTM recurrence. 16 WGs x 256 threads (8 waves).
// WG w owns hidden units j in [32w, 32w+32): computes gate columns
// {g*512 + 32w .. +32} for g=0..3 so its cell state c stays in registers.
// Per step: M=64 x (its 128 gate cols) x K=512 bf16 GEMM (software-pipelined
// by one K-tile), then activations. h double-buffered in global bf16; one
// grid barrier per timestep.
// ---------------------------------------------------------------------------
__global__ __launch_bounds__(256) void lstm_kernel(
    const unsigned short* __restrict__ px,     // [B*S][2048] bf16
    const unsigned short* __restrict__ WhhP,   // packed B tiles (128 ntiles, 16 ktiles)
    const float* __restrict__ b_hh,            // [2048]
    unsigned short* __restrict__ hs,           // [B*S][512] bf16 out
    unsigned short* __restrict__ hbuf,         // [2][64][512] bf16
    unsigned* __restrict__ cnt) {
  const unsigned w    = blockIdx.x;
  const unsigned j0   = w * 32u;
  const unsigned tid  = threadIdx.x;
  const unsigned wv   = tid >> 5;
  const unsigned lane = tid & 31u;
  const unsigned mtile = wv & 3u;        // batch tile (4 x 16 rows = 64)
  const unsigned nh    = wv >> 2;        // which 4 of the 8 N-tiles
  const unsigned m0    = mtile * 16u;
  const unsigned half  = lane >> 4, n15 = lane & 15u;

  __shared__ float gbuf[64][132];  // gates slice [batch][4*32], padded

  // B-tile base pointers: 4 per wave, fixed for the whole sequence.
  const unsigned short* bB[4];
#pragma unroll
  for (int q = 0; q < 4; ++q) {
    const unsigned qq    = nh * 4u + q;            // 0..7
    const unsigned gate  = qq >> 1;
    const unsigned ntile = gate * 32u + (j0 >> 4) + (qq & 1u);
    bB[q] = b_base(WhhP, 16, ntile, lane);
    __builtin_prefetch(bB[q], 0, 3);
  }

  // zero h buffer 0 for our hidden slice
  for (unsigned i = tid; i < 64u * 32u; i += 256u) {
    unsigned b = i >> 5, jj = i & 31u;
    hbuf[b * H_ + j0 + jj] = 0;
  }
  unsigned expected = NWG_LSTM;
  grid_barrier(cnt, expected);

  float c[8];
#pragma unroll
  for (int k = 0; k < 8; ++k) c[k] = 0.0f;

  for (unsigned t = 0; t < S_; ++t) {
    const unsigned short* hr = hbuf + (size_t)(t & 1u) * B_ * H_;
    unsigned short*       hw = hbuf + (size_t)((t + 1u) & 1u) * B_ * H_;
    const unsigned short* aB = a_base(hr, H_, m0, lane);

    v16bf a0 = frag2(aB, 0u, 16u);
    v16bf b0[4];
#pragma unroll
    for (int q = 0; q < 4; ++q) b0[q] = frag2(bB[q], 0u, 8u);

    v8f acc[4] = {};
#pragma unroll
    for (unsigned kt = 0; kt < 16; ++kt) {        // K = 512, all-immediate offsets
      v16bf a1 = a0;
      v16bf b1[4];
#pragma unroll
      for (int q = 0; q < 4; ++q) b1[q] = b0[q];
      if (kt < 15) {                               // issue next tile's loads first
        a1 = frag2(aB, (kt + 1) * 32u, (kt + 1) * 32u + 16u);
#pragma unroll
        for (int q = 0; q < 4; ++q) b1[q] = frag2(bB[q], (kt + 1) * 512u, (kt + 1) * 512u + 8u);
      }
#pragma unroll
      for (int q = 0; q < 4; ++q) acc[q] = wmma_bf16(a0, b0[q], acc[q]);
      a0 = a1;
#pragma unroll
      for (int q = 0; q < 4; ++q) b0[q] = b1[q];
    }
    // dump accumulators to LDS: col = qq*16 + n15 == gate*32 + jj
#pragma unroll
    for (int q = 0; q < 4; ++q) {
      const unsigned qq = nh * 4u + q;
#pragma unroll
      for (int r = 0; r < 8; ++r)
        gbuf[m0 + half * 8u + r][qq * 16u + n15] = acc[q][r];
    }
    __syncthreads();

    // elementwise gates: each thread owns 8 (b, jj) cells for all 512 steps
#pragma unroll
    for (int k = 0; k < 8; ++k) {
      const unsigned cell = tid + k * 256u;
      const unsigned b  = cell >> 5;
      const unsigned jj = cell & 31u;
      const unsigned gj = j0 + jj;
      const size_t prow = ((size_t)b * S_ + t) * G_;
      float xi = bf2f(px[prow + 0 * H_ + gj]) + gbuf[b][ 0 + jj] + b_hh[0 * H_ + gj];
      float xf = bf2f(px[prow + 1 * H_ + gj]) + gbuf[b][32 + jj] + b_hh[1 * H_ + gj];
      float xg = bf2f(px[prow + 2 * H_ + gj]) + gbuf[b][64 + jj] + b_hh[2 * H_ + gj];
      float xo = bf2f(px[prow + 3 * H_ + gj]) + gbuf[b][96 + jj] + b_hh[3 * H_ + gj];
      float ig = sigf(xi), fg = sigf(xf), gg = tanhf(xg), og = sigf(xo);
      c[k] = fg * c[k] + ig * gg;
      float h = og * tanhf(c[k]);
      unsigned short hb = f2bf(h);
      hw[b * H_ + gj] = hb;
      hs[((size_t)b * S_ + t) * H_ + gj] = hb;
    }
    grid_barrier(cnt, expected);   // includes __syncthreads(): protects gbuf reuse
  }
}

// ---------------------------------------------------------------------------
// Kernel: logits = hs @ fc_W^T + fc_b, then log_softmax over 48 tags.
// WG: 8 waves x 16 rows = 128 rows; 3 N-tiles (48 cols). Grid 256.
// ---------------------------------------------------------------------------
__global__ __launch_bounds__(256) void fc_softmax_kernel(
    const unsigned short* __restrict__ hs,
    const unsigned short* __restrict__ fcWP,
    const float* __restrict__ fc_b,
    float* __restrict__ out) {
  const unsigned wv   = threadIdx.x >> 5;
  const unsigned lane = threadIdx.x & 31u;
  const unsigned m0   = blockIdx.x * 128u + wv * 16u;
  const unsigned half = lane >> 4, n15 = lane & 15u;

  __shared__ float lg[128][52];

  const unsigned short* aB = a_base(hs, H_, m0, lane);
  const unsigned short* bB[3];
#pragma unroll
  for (int j = 0; j < 3; ++j) bB[j] = b_base(fcWP, 16, j, lane);

  v16bf a0 = frag2(aB, 0u, 16u);
  v16bf b0[3];
#pragma unroll
  for (int j = 0; j < 3; ++j) b0[j] = frag2(bB[j], 0u, 8u);

  v8f acc[3] = {};
#pragma unroll
  for (unsigned kt = 0; kt < 16; ++kt) {          // K = 512
    v16bf a1 = a0;
    v16bf b1[3];
#pragma unroll
    for (int j = 0; j < 3; ++j) b1[j] = b0[j];
    if (kt < 15) {
      a1 = frag2(aB, (kt + 1) * 32u, (kt + 1) * 32u + 16u);
#pragma unroll
      for (int j = 0; j < 3; ++j) b1[j] = frag2(bB[j], (kt + 1) * 512u, (kt + 1) * 512u + 8u);
    }
#pragma unroll
    for (int j = 0; j < 3; ++j) acc[j] = wmma_bf16(a0, b0[j], acc[j]);
    a0 = a1;
#pragma unroll
    for (int j = 0; j < 3; ++j) b0[j] = b1[j];
  }
#pragma unroll
  for (int j = 0; j < 3; ++j) {
    const unsigned col = j * 16u + n15;
    const float bb = fc_b[col];
#pragma unroll
    for (int r = 0; r < 8; ++r)
      lg[wv * 16u + half * 8u + r][col] = acc[j][r] + bb;
  }
  __syncthreads();

  if (threadIdx.x < 128) {
    const unsigned row = threadIdx.x;
    float mx = -3.4e38f;
#pragma unroll
    for (int cix = 0; cix < T_; ++cix) mx = fmaxf(mx, lg[row][cix]);
    float s = 0.0f;
#pragma unroll
    for (int cix = 0; cix < T_; ++cix) s += expf(lg[row][cix] - mx);
    const float lse = mx + logf(s);
    const size_t m = (size_t)blockIdx.x * 128u + row;
#pragma unroll
    for (int cix = 0; cix < T_; ++cix) out[m * T_ + cix] = lg[row][cix] - lse;
  }
}

// ---------------------------------------------------------------------------
// Host launcher
// ---------------------------------------------------------------------------
extern "C" void kernel_launch(void* const* d_in, const int* in_sizes, int n_in,
                              void* d_out, int out_size, void* d_ws, size_t ws_size,
                              hipStream_t stream) {
  const int*   x    = (const int*)d_in[0];
  const float* emb  = (const float*)d_in[1];
  const float* W_ih = (const float*)d_in[2];
  const float* W_hh = (const float*)d_in[3];
  const float* b_hh = (const float*)d_in[4];
  const float* fc_W = (const float*)d_in[5];
  const float* fc_b = (const float*)d_in[6];
  float* out = (float*)d_out;

  char* ws = (char*)d_ws;
  size_t off = 0;
  auto alloc = [&](size_t bytes) { char* p = ws + off; off = (off + bytes + 255) & ~(size_t)255; return p; };
  unsigned short* WihP = (unsigned short*)alloc((size_t)G_ * E_ * 2);        //  1 MB
  unsigned short* WhhP = (unsigned short*)alloc((size_t)G_ * H_ * 2);        //  2 MB
  unsigned short* fcWP = (unsigned short*)alloc((size_t)T_ * H_ * 2);        // 48 KB
  unsigned short* xe   = (unsigned short*)alloc((size_t)B_ * S_ * E_ * 2);   // 16 MB
  unsigned short* px   = (unsigned short*)alloc((size_t)B_ * S_ * G_ * 2);   // 134 MB
  unsigned short* hs   = (unsigned short*)alloc((size_t)B_ * S_ * H_ * 2);   // 33.5 MB
  unsigned short* hbuf = (unsigned short*)alloc((size_t)2 * B_ * H_ * 2);    // 128 KB
  unsigned*       cnt  = (unsigned*)alloc(256);

  hipMemsetAsync(cnt, 0, 256, stream);

  pack_b_kernel<<<(G_ * E_ + 255) / 256, 256, 0, stream>>>(W_ih, WihP, G_, E_);
  pack_b_kernel<<<(G_ * H_ + 255) / 256, 256, 0, stream>>>(W_hh, WhhP, G_, H_);
  pack_b_kernel<<<(T_ * H_ + 255) / 256, 256, 0, stream>>>(fc_W, fcWP, T_, H_);

  embed_kernel<<<(B_ * S_ * E_) / 256, 256, 0, stream>>>(x, emb, xe);

  gemm_px_kernel<<<dim3((B_ * S_) / 128, G_ / 64), 256, 0, stream>>>(xe, WihP, px);

  lstm_kernel<<<NWG_LSTM, 256, 0, stream>>>(px, WhhP, b_hh, hs, hbuf, cnt);

  fc_softmax_kernel<<<(B_ * S_) / 128, 256, 0, stream>>>(hs, fcWP, fc_b, out);
}
